// PointNetSA_25735444037745
// MI455X (gfx1250) — matrix operand order repaired and checked by reading, not compiled, not run on hardware
//
#include <hip/hip_runtime.h>
#include <hip/hip_bf16.h>

typedef __attribute__((ext_vector_type(16))) _Float16 v16h;
typedef __attribute__((ext_vector_type(8)))  _Float16 v8h;
typedef __attribute__((ext_vector_type(8)))  float    v8f;

#define NB    8
#define NPTS  8192
#define NF    64
#define SQ    2048
#define NSAMP 32
#define R2    0.04f
#define SLOPE 0.2f
#define EPSV  1e-5f

// ---------------------------------------------------------------------------
// lane helpers (wave32)
// ---------------------------------------------------------------------------
__device__ __forceinline__ int lane_id() { return threadIdx.x & 31; }

__device__ __forceinline__ int shflx_i(int x, int m) {
    int idx = ((lane_id() ^ m) & 31) << 2;
    return __builtin_amdgcn_ds_bpermute(idx, x);
}
__device__ __forceinline__ float shflx_f(float x, int m) {
    return __builtin_bit_cast(float, shflx_i(__builtin_bit_cast(int, x), m));
}

// ---------------------------------------------------------------------------
// Kernel 1: fold BN scale into weights, emit f16 WMMA B-fragments.
// Fragment storage: frag f, half index h = lane*16 + e.
// B layout (16x16x32 f16): lane L -> col N = L&15, K = k*32 + ((L>>4)<<3) + (e&7) + ((e>>3)<<4)
// Fragment order: L1 n*3+k (12), L2 12+n*2+k (8), L3 20+n*2+k (16).
// ---------------------------------------------------------------------------
__global__ void wprep_kernel(const float* __restrict__ w1, const float* __restrict__ g1, const float* __restrict__ v1,
                             const float* __restrict__ w2, const float* __restrict__ g2, const float* __restrict__ v2,
                             const float* __restrict__ w3, const float* __restrict__ g3, const float* __restrict__ v3,
                             _Float16* __restrict__ wfrag) {
    int f = blockIdx.x;          // 0..35
    int h = threadIdx.x;         // 0..511
    int lane = h >> 4;
    int e    = h & 15;
    const float *w, *g, *vv;
    int n, k, cin;
    if (f < 12)      { n = f / 3;        k = f % 3;        cin = 67; w = w1; g = g1; vv = v1; }
    else if (f < 20) { int i = f - 12; n = i / 2; k = i % 2; cin = 64; w = w2; g = g2; vv = v2; }
    else             { int i = f - 20; n = i / 2; k = i % 2; cin = 64; w = w3; g = g3; vv = v3; }
    int o = n * 16 + (lane & 15);
    int c = k * 32 + ((lane >> 4) << 3) + (e & 7) + ((e >> 3) << 4);
    float val = 0.0f;
    if (c < cin) val = w[o * cin + c] * (g[o] * rsqrtf(vv[o] + EPSV));
    wfrag[f * 512 + h] = (_Float16)val;
}

// ---------------------------------------------------------------------------
// Kernel 2: fused biases  (b - m)*g*rsqrt(v+eps) + bb   (64 + 64 + 128)
// ---------------------------------------------------------------------------
__global__ void bprep_kernel(const float* b1, const float* g1, const float* bb1, const float* m1, const float* v1,
                             const float* b2, const float* g2, const float* bb2, const float* m2, const float* v2,
                             const float* b3, const float* g3, const float* bb3, const float* m3, const float* v3,
                             float* __restrict__ bias) {
    int o = threadIdx.x;  // 0..255
    float r;
    if (o < 64) {
        float s = g1[o] * rsqrtf(v1[o] + EPSV);
        r = (b1[o] - m1[o]) * s + bb1[o];
    } else if (o < 128) {
        int i = o - 64;
        float s = g2[i] * rsqrtf(v2[i] + EPSV);
        r = (b2[i] - m2[i]) * s + bb2[i];
    } else {
        int i = o - 128;
        float s = g3[i] * rsqrtf(v3[i] + EPSV);
        r = (b3[i] - m3[i]) * s + bb3[i];
    }
    bias[o] = r;
}

// ---------------------------------------------------------------------------
// Kernel 3: furthest point sampling.  One block (1024 thr) per batch.
// Matches jnp.argmax tie-break: max value, then lowest index.
// Writes query coords (ws) and new_xyz (d_out tail).
// ---------------------------------------------------------------------------
__global__ __launch_bounds__(1024, 1)
void fps_kernel(const float* __restrict__ points, float* __restrict__ qarr,
                float* __restrict__ out_xyz) {
    int bq  = blockIdx.x;
    int tid = threadIdx.x;
    int lane = tid & 31, w = tid >> 5;
    const float* px = points + (size_t)bq * 3 * NPTS;
    const float* py = px + NPTS;
    const float* pz = py + NPTS;

    float dist[8];
#pragma unroll
    for (int j = 0; j < 8; ++j) dist[j] = 1e10f;

    __shared__ float sval[32];
    __shared__ int   sidx[32];
    __shared__ int   sfar;

    int far = 0;
    for (int it = 0; it < SQ; ++it) {
        float cx = px[far], cy = py[far], cz = pz[far];
        if (tid == 0) {
            int q = bq * SQ + it;
            qarr[q * 3 + 0] = cx; qarr[q * 3 + 1] = cy; qarr[q * 3 + 2] = cz;
            out_xyz[(size_t)bq * 3 * SQ + 0 * SQ + it] = cx;
            out_xyz[(size_t)bq * 3 * SQ + 1 * SQ + it] = cy;
            out_xyz[(size_t)bq * 3 * SQ + 2 * SQ + it] = cz;
        }
        float bv = -1.0f; int bi = 0;
#pragma unroll
        for (int j = 0; j < 8; ++j) {
            int n = j * 1024 + tid;
            float dx = px[n] - cx, dy = py[n] - cy, dz = pz[n] - cz;
            float d = dx * dx + dy * dy + dz * dz;
            float nd = fminf(dist[j], d);
            dist[j] = nd;
            if (nd > bv) { bv = nd; bi = n; }
        }
        // wave-level lexicographic (max val, min idx) reduction
#pragma unroll
        for (int off = 16; off > 0; off >>= 1) {
            float ov = shflx_f(bv, off);
            int   oi = shflx_i(bi, off);
            if (ov > bv || (ov == bv && oi < bi)) { bv = ov; bi = oi; }
        }
        if (lane == 0) { sval[w] = bv; sidx[w] = bi; }
        __syncthreads();
        if (w == 0) {
            bv = sval[lane]; bi = sidx[lane];
#pragma unroll
            for (int off = 16; off > 0; off >>= 1) {
                float ov = shflx_f(bv, off);
                int   oi = shflx_i(bi, off);
                if (ov > bv || (ov == bv && oi < bi)) { bv = ov; bi = oi; }
            }
            if (lane == 0) sfar = bi;
        }
        __syncthreads();
        far = sfar;
    }
}

// ---------------------------------------------------------------------------
// Kernel 4: fused ball-query + gather + 3-layer WMMA MLP + max-pool.
// One wave per query; 2048 blocks x 8 waves = 16384 waves.
// A layout (16x16x32 f16): lane L -> row M = L&15,
//   K = kchunk*32 + ((L>>4)<<3) + (e&7) + ((e>>3)<<4)
// C/D layout: lane L -> col N = L&15, VGPR r -> row M = r + 8*(L>>4)
// B fragments streamed from global (36 KB total, L2-resident) to keep
// VGPR pressure < 256 (avoids s_set_vgpr_msb churn, better occupancy).
// ---------------------------------------------------------------------------
__device__ __forceinline__ v8f wmma_f16(v16h a, v16h b, v8f c) {
    return __builtin_amdgcn_wmma_f32_16x16x32_f16(false, a, false, b, (short)0, c, false, false);
}

__global__ __launch_bounds__(256, 1)
void group_mlp_kernel(const float* __restrict__ points, const float* __restrict__ features,
                      const _Float16* __restrict__ wfrag, const float* __restrict__ bias,
                      const float* __restrict__ qarr, float* __restrict__ out) {
    __shared__ int idx_s[8][32];
    __shared__ __align__(16) _Float16 xbuf[8][32 * 64];

    int lane = threadIdx.x & 31;
    int w    = threadIdx.x >> 5;
    int qid  = blockIdx.x * 8 + w;       // 0..16383 : one query per wave
    int hi   = lane >> 4;                // lane half
    int l15  = lane & 15;

    float bi1[4], bi2[4], bi3[8];
#pragma unroll
    for (int n = 0; n < 4; ++n) { bi1[n] = bias[n * 16 + l15]; bi2[n] = bias[64 + n * 16 + l15]; }
#pragma unroll
    for (int n = 0; n < 8; ++n) bi3[n] = bias[128 + n * 16 + l15];

    int bq = qid >> 11;
    int s  = qid & (SQ - 1);
    float qx = qarr[qid * 3 + 0], qy = qarr[qid * 3 + 1], qz = qarr[qid * 3 + 2];
    const float* px = points + (size_t)bq * 3 * NPTS;
    const float* py = px + NPTS;
    const float* pz = py + NPTS;

    // ---- ball query: first 32 in-ball indices, ascending ----
    int cnt = 0;
    for (int base = 0; base < NPTS && cnt < NSAMP; base += 32) {
        int n = base + lane;
        float dx = px[n] - qx, dy = py[n] - qy, dz = pz[n] - qz;
        bool inb = (dx * dx + dy * dy + dz * dz) < R2;
        unsigned m = __builtin_amdgcn_ballot_w32(inb);
        int slot = cnt + __popc(m & ((1u << lane) - 1u));
        if (inb && slot < NSAMP) idx_s[w][slot] = n;
        cnt += __popc(m);
    }
    if (cnt > NSAMP) cnt = NSAMP;
    int first = idx_s[w][0];             // query itself is always in-ball
    if (lane >= cnt) idx_s[w][lane] = first;

    // ---- layer 1: per m-tile, gather A fragments then 4 WMMA col-tiles ----
#pragma unroll
    for (int m = 0; m < 2; ++m) {
        int si = idx_s[w][m * 16 + l15];
        float gx = px[si] - qx, gy = py[si] - qy, gz = pz[si] - qz;
        const float* fb = features + (size_t)bq * NF * NPTS + si;
        v16h a1k[3];
#pragma unroll
        for (int k = 0; k < 3; ++k) {
            v16h a;
#pragma unroll
            for (int e = 0; e < 16; ++e) {
                int c = k * 32 + (hi << 3) + (e & 7) + ((e >> 3) << 4);
                float val;
                if (c == 0)      val = gx;
                else if (c == 1) val = gy;
                else if (c == 2) val = gz;
                else if (c < 67) val = fb[(size_t)(c - 3) * NPTS];
                else             val = 0.0f;
                a[e] = (_Float16)val;
            }
            a1k[k] = a;
        }
#pragma unroll
        for (int n = 0; n < 4; ++n) {
            v8f acc; float bv = bi1[n];
#pragma unroll
            for (int i = 0; i < 8; ++i) acc[i] = bv;
#pragma unroll
            for (int k = 0; k < 3; ++k) {
                v16h bf = *(const v16h*)(wfrag + (n * 3 + k) * 512 + lane * 16);
                acc = wmma_f16(a1k[k], bf, acc);
            }
#pragma unroll
            for (int r = 0; r < 8; ++r) {
                float y = acc[r];
                y = y > 0.0f ? y : SLOPE * y;
                xbuf[w][(m * 16 + r + (hi << 3)) * 64 + n * 16 + l15] = (_Float16)y;
            }
        }
    }

    // ---- layer 2: per m-tile (m=0 writes rows 0-15, m=1 reads rows 16-31:
    //      disjoint, so in-place xbuf reuse is safe) ----
#pragma unroll
    for (int m = 0; m < 2; ++m) {
        int srow = m * 16 + l15;
        v16h a2k[2];
#pragma unroll
        for (int k = 0; k < 2; ++k) {
            int c0 = k * 32 + (hi << 3);
            v8h lo = *(const v8h*)&xbuf[w][srow * 64 + c0];
            v8h hh = *(const v8h*)&xbuf[w][srow * 64 + c0 + 16];
            a2k[k] = __builtin_shufflevector(lo, hh, 0,1,2,3,4,5,6,7,8,9,10,11,12,13,14,15);
        }
#pragma unroll
        for (int n = 0; n < 4; ++n) {
            v8f acc; float bv = bi2[n];
#pragma unroll
            for (int i = 0; i < 8; ++i) acc[i] = bv;
#pragma unroll
            for (int k = 0; k < 2; ++k) {
                v16h bf = *(const v16h*)(wfrag + (12 + n * 2 + k) * 512 + lane * 16);
                acc = wmma_f16(a2k[k], bf, acc);
            }
#pragma unroll
            for (int r = 0; r < 8; ++r) {
                float y = acc[r];
                y = y > 0.0f ? y : SLOPE * y;
                xbuf[w][(m * 16 + r + (hi << 3)) * 64 + n * 16 + l15] = (_Float16)y;
            }
        }
    }

    // ---- layer 3 A fragments (both m-tiles) ----
    v16h a3[2][2];
#pragma unroll
    for (int m = 0; m < 2; ++m)
#pragma unroll
        for (int k = 0; k < 2; ++k) {
            int srow = m * 16 + l15;
            int c0 = k * 32 + (hi << 3);
            v8h lo = *(const v8h*)&xbuf[w][srow * 64 + c0];
            v8h hh = *(const v8h*)&xbuf[w][srow * 64 + c0 + 16];
            a3[m][k] = __builtin_shufflevector(lo, hh, 0,1,2,3,4,5,6,7,8,9,10,11,12,13,14,15);
        }

    // ---- layer 3: (32x64) x (64x128) + max-pool over 32 samples ----
#pragma unroll
    for (int n = 0; n < 8; ++n) {
        float pm = -3.4e38f;
#pragma unroll
        for (int m = 0; m < 2; ++m) {
            v8f acc; float bv = bi3[n];
#pragma unroll
            for (int i = 0; i < 8; ++i) acc[i] = bv;
#pragma unroll
            for (int k = 0; k < 2; ++k) {
                v16h bf = *(const v16h*)(wfrag + (20 + n * 2 + k) * 512 + lane * 16);
                acc = wmma_f16(a3[m][k], bf, acc);
            }
#pragma unroll
            for (int r = 0; r < 8; ++r) {
                float y = acc[r];
                y = y > 0.0f ? y : SLOPE * y;
                pm = fmaxf(pm, y);
            }
        }
        pm = fmaxf(pm, shflx_f(pm, 16));   // combine lane halves (other 16 rows)
        if (lane < 16)
            out[((size_t)bq * 128 + n * 16 + lane) * SQ + s] = pm;
    }
}

// ---------------------------------------------------------------------------
// host launcher
// ---------------------------------------------------------------------------
extern "C" void kernel_launch(void* const* d_in, const int* in_sizes, int n_in,
                              void* d_out, int out_size, void* d_ws, size_t ws_size,
                              hipStream_t stream) {
    (void)in_sizes; (void)n_in; (void)out_size; (void)ws_size;
    const float* features = (const float*)d_in[0];
    const float* points   = (const float*)d_in[1];
    const float* w1 = (const float*)d_in[2],  *b1 = (const float*)d_in[3],
               * g1 = (const float*)d_in[4],  *bb1 = (const float*)d_in[5],
               * m1 = (const float*)d_in[6],  *v1 = (const float*)d_in[7];
    const float* w2 = (const float*)d_in[8],  *b2 = (const float*)d_in[9],
               * g2 = (const float*)d_in[10], *bb2 = (const float*)d_in[11],
               * m2 = (const float*)d_in[12], *v2 = (const float*)d_in[13];
    const float* w3 = (const float*)d_in[14], *b3 = (const float*)d_in[15],
               * g3 = (const float*)d_in[16], *bb3 = (const float*)d_in[17],
               * m3 = (const float*)d_in[18], *v3 = (const float*)d_in[19];

    char* ws = (char*)d_ws;
    _Float16* wfrag = (_Float16*)ws;                 // 36 * 512 * 2  = 36864 B
    float*    bias  = (float*)(ws + 36864);          // 256 * 4       =  1024 B
    float*    qarr  = (float*)(ws + 40960);          // 8*2048*3*4    = 196608 B

    float* out      = (float*)d_out;
    float* out_xyz  = out + (size_t)NB * 128 * SQ;

    wprep_kernel<<<36, 512, 0, stream>>>(w1, g1, v1, w2, g2, v2, w3, g3, v3, wfrag);
    bprep_kernel<<<1, 256, 0, stream>>>(b1, g1, bb1, m1, v1, b2, g2, bb2, m2, v2,
                                        b3, g3, bb3, m3, v3, bias);
    fps_kernel<<<NB, 1024, 0, stream>>>(points, qarr, out_xyz);
    group_mlp_kernel<<<2048, 256, 0, stream>>>(points, features, wfrag, bias, qarr, out);
}
